// FlowEmbedding_78426102825592
// MI455X (gfx1250) — matrix operand order repaired and checked by reading, not compiled, not run on hardware
//
#include <hip/hip_runtime.h>
#include <cstdint>
#include <cstddef>

#define Bsz   2
#define Npts  8192
#define Cfeat 128
#define KNN   16
#define NKc   131072   // Npts*KNN = 2^17
#define KP1   288      // 259 padded to multiple of 32
#define CIN1  259
#define NGRP  8        // 128 / 16 channels per group
#define EPSV  1e-5f
#define SLOPE 0.1f
#define BSTR  40       // LDS B-tile column stride (shorts): 80B, 16B-aligned

typedef __attribute__((ext_vector_type(16))) __bf16 v16bf;
typedef __attribute__((ext_vector_type(8)))  float  v8f;

union Frag { v16bf v; uint4 q[2]; };

__device__ __forceinline__ unsigned short f2bf(float f) {
  unsigned int u = __float_as_uint(f);
  u += 0x7FFFu + ((u >> 16) & 1u);          // round-to-nearest-even
  return (unsigned short)(u >> 16);
}

// ---------------------------------------------------------------------------
// 1) Brute-force KNN: one thread per query point, LDS-tiled candidates,
//    branchless 16-deep sorted insert kept in registers.
// ---------------------------------------------------------------------------
__global__ void knn_kernel(const float* __restrict__ xyz1,
                           const float* __restrict__ xyz2,
                           int* __restrict__ idxOut) {
  __shared__ float lx[128], ly[128], lz[128];
  const int b = blockIdx.y;
  const int q = blockIdx.x * 128 + threadIdx.x;
  const float qx = xyz1[(b * 3 + 0) * Npts + q];
  const float qy = xyz1[(b * 3 + 1) * Npts + q];
  const float qz = xyz1[(b * 3 + 2) * Npts + q];

  float bd[KNN];
  int   bi[KNN];
#pragma unroll
  for (int p = 0; p < KNN; ++p) { bd[p] = 3.4e38f; bi[p] = 0; }

  for (int j0 = 0; j0 < Npts; j0 += 128) {
    __syncthreads();
    const int t = threadIdx.x;
    lx[t] = xyz2[(b * 3 + 0) * Npts + j0 + t];
    ly[t] = xyz2[(b * 3 + 1) * Npts + j0 + t];
    lz[t] = xyz2[(b * 3 + 2) * Npts + j0 + t];
    __syncthreads();
    for (int jj = 0; jj < 128; ++jj) {
      const float dx = qx - lx[jj];
      const float dy = qy - ly[jj];
      const float dz = qz - lz[jj];
      float d = dx * dx + dy * dy + dz * dz;
      int   id = j0 + jj;
#pragma unroll
      for (int p = 0; p < KNN; ++p) {
        const bool lt = d < bd[p];
        const float tb = bd[p]; const int ti = bi[p];
        bd[p] = lt ? d : tb;  bi[p] = lt ? id : ti;
        d     = lt ? tb : d;  id    = lt ? ti : id;
      }
    }
  }
#pragma unroll
  for (int p = 0; p < KNN; ++p)
    idxOut[((size_t)b * Npts + q) * KNN + p] = bi[p];
}

// ---------------------------------------------------------------------------
// 2) Build bf16 GEMM input X0 [b][ci(0..287)][col = n*16+k], zero-padded rows.
// ---------------------------------------------------------------------------
__global__ void build_input_kernel(const float* __restrict__ feat1,
                                   const float* __restrict__ feat2,
                                   const float* __restrict__ xyz1,
                                   const float* __restrict__ xyz2,
                                   const int* __restrict__ idx,
                                   unsigned short* __restrict__ X) {
  const unsigned int tid = blockIdx.x * 256u + threadIdx.x;
  if (tid >= (unsigned)Bsz * KP1 * NKc) return;
  const int b   = tid / (KP1 * NKc);
  const int ci  = (tid >> 17) % KP1;
  const int col = tid & (NKc - 1);
  const int n   = col >> 4;
  const int k   = col & 15;
  float v = 0.f;
  if (ci < 128) {
    v = feat1[((size_t)b * 128 + ci) * Npts + n];
  } else if (ci < 256) {
    const int j = idx[((size_t)b * Npts + n) * KNN + k];
    v = feat2[((size_t)b * 128 + (ci - 128)) * Npts + j];
  } else if (ci < CIN1) {
    const int d = ci - 256;
    const int j = idx[((size_t)b * Npts + n) * KNN + k];
    v = xyz2[((size_t)b * 3 + d) * Npts + j] - xyz1[((size_t)b * 3 + d) * Npts + n];
  }
  X[tid] = f2bf(v);
}

// ---------------------------------------------------------------------------
// 3) Convert fp32 weights to bf16, K-padded with zeros.
// ---------------------------------------------------------------------------
__global__ void wconv_kernel(const float* __restrict__ W,
                             unsigned short* __restrict__ Wbf,
                             int Cin, int Kpad) {
  const int tid = blockIdx.x * 256 + threadIdx.x;
  if (tid >= 128 * Kpad) return;
  const int c = tid / Kpad, k = tid % Kpad;
  Wbf[tid] = (k < Cin) ? f2bf(W[c * Cin + k]) : (unsigned short)0;
}

// ---------------------------------------------------------------------------
// 4) WMMA GEMM: Y[b][m][col] = sum_k Wbf[m][k] * X[b][k][col] + bias[m]
//    WG tile M=128 x N=64, 8 waves. Double-buffered LDS, software-pipelined:
//    next tile's global_load_b128 issue before the barrier (HBM latency hides
//    under WMMAs), LDS stores after the WMMAs into the alternate buffer.
//    All fragment ds_load_b128 grouped before the 4 independent WMMAs.
// ---------------------------------------------------------------------------
__global__ void gemm_gn_kernel(const unsigned short* __restrict__ Wbf,
                               const unsigned short* __restrict__ X,
                               const float* __restrict__ bias,
                               float* __restrict__ Y,
                               float* __restrict__ stats,
                               int Kpad) {
  __shared__ __align__(16) unsigned short ldsA[2][128 * 32];   // [m][k] 2x8KB
  __shared__ __align__(16) unsigned short ldsBt[2][64 * BSTR]; // [col][k] 2x5KB
  __shared__ float ldsBias[128];
  __shared__ float ldsSum[NGRP], ldsSq[NGRP];

  const int t    = threadIdx.x;
  const int wave = t >> 5;
  const int lane = t & 31;
  const int hi   = lane >> 4;
  const int b    = blockIdx.y;
  const int colBase = blockIdx.x * 64;
  const unsigned short* Xb = X + (size_t)b * Kpad * NKc;

  if (t < 128) ldsBias[t] = bias[t];
  if (t < NGRP) { ldsSum[t] = 0.f; ldsSq[t] = 0.f; }

  const v8f vzero = {0.f, 0.f, 0.f, 0.f, 0.f, 0.f, 0.f, 0.f};
  v8f acc[4];
#pragma unroll
  for (int nt = 0; nt < 4; ++nt) acc[nt] = vzero;

  const int Mtile = wave * 16;
  const int aRow  = (Mtile + (lane & 15)) * 32;
  // staging thread roles
  const int sRow0 = t >> 2, sU = t & 3;          // A: rows 0..63  (it=0)
  const int sRow1 = (t + 256) >> 2;              // A: rows 64..127 (it=1)
  const int krow  = t >> 3;                      // B: k row 0..31
  const int c0    = (t & 7) * 8;                 // B: col group

  uint4 aw0, aw1, bw;                            // in-flight staging registers
  auto GLOAD = [&](int kk) {
    aw0 = *(const uint4*)(Wbf + (size_t)sRow0 * Kpad + kk + sU * 8);
    aw1 = *(const uint4*)(Wbf + (size_t)sRow1 * Kpad + kk + sU * 8);
    bw  = *(const uint4*)(Xb + (size_t)(kk + krow) * NKc + colBase + c0);
  };
  auto SSTORE = [&](int buf) {
    *(uint4*)&ldsA[buf][sRow0 * 32 + sU * 8] = aw0;
    *(uint4*)&ldsA[buf][sRow1 * 32 + sU * 8] = aw1;
    const unsigned int* dw = (const unsigned int*)&bw;
#pragma unroll
    for (int i = 0; i < 4; ++i) {
      ldsBt[buf][(c0 + 2 * i)     * BSTR + krow] = (unsigned short)(dw[i] & 0xFFFFu);
      ldsBt[buf][(c0 + 2 * i + 1) * BSTR + krow] = (unsigned short)(dw[i] >> 16);
    }
  };

  GLOAD(0);
  SSTORE(0);

  int buf = 0;
  for (int kk = 0; kk < Kpad; kk += 32, buf ^= 1) {
    const bool more = (kk + 32) < Kpad;
    if (more) GLOAD(kk + 32);                    // HBM loads fly under compute
    if (kk + 64 < Kpad)
      __builtin_prefetch(Xb + (size_t)(kk + 64) * NKc + colBase, 0, 0);
    __syncthreads();                             // buf's stores visible

    // all fragment loads grouped -> one wait, then 4 independent WMMAs
    Frag a, bf0, bf1, bf2, bf3;
    a.q[0] = *(const uint4*)&ldsA[buf][aRow + hi * 8];
    a.q[1] = *(const uint4*)&ldsA[buf][aRow + 16 + hi * 8];
    {
      const unsigned short* bp = &ldsBt[buf][(lane & 15) * BSTR + hi * 8];
      bf0.q[0] = *(const uint4*)bp;        bf0.q[1] = *(const uint4*)(bp + 16);
      bf1.q[0] = *(const uint4*)(bp + 16 * BSTR);
      bf1.q[1] = *(const uint4*)(bp + 16 * BSTR + 16);
      bf2.q[0] = *(const uint4*)(bp + 32 * BSTR);
      bf2.q[1] = *(const uint4*)(bp + 32 * BSTR + 16);
      bf3.q[0] = *(const uint4*)(bp + 48 * BSTR);
      bf3.q[1] = *(const uint4*)(bp + 48 * BSTR + 16);
    }
    acc[0] = __builtin_amdgcn_wmma_f32_16x16x32_bf16(
        false, a.v, false, bf0.v, (short)0, acc[0], false, false);
    acc[1] = __builtin_amdgcn_wmma_f32_16x16x32_bf16(
        false, a.v, false, bf1.v, (short)0, acc[1], false, false);
    acc[2] = __builtin_amdgcn_wmma_f32_16x16x32_bf16(
        false, a.v, false, bf2.v, (short)0, acc[2], false, false);
    acc[3] = __builtin_amdgcn_wmma_f32_16x16x32_bf16(
        false, a.v, false, bf3.v, (short)0, acc[3], false, false);

    if (more) SSTORE(buf ^ 1);                   // store next tile after compute
  }

  // epilogue: bias, store fp32, per-group stats (group == wave, M-tile aligned)
  float s = 0.f, sq = 0.f;
#pragma unroll
  for (int nt = 0; nt < 4; ++nt) {
#pragma unroll
    for (int r = 0; r < 8; ++r) {
      const int m = Mtile + r + (hi << 3);
      const int n = colBase + nt * 16 + (lane & 15);
      const float v = acc[nt][r] + ldsBias[m];
      Y[((size_t)(b * 128 + m)) * NKc + n] = v;
      s += v; sq += v * v;
    }
  }
  atomicAdd(&ldsSum[wave], s);
  atomicAdd(&ldsSq[wave], sq);
  __syncthreads();
  if (t < NGRP) {
    atomicAdd(&stats[(b * NGRP + t) * 2 + 0], ldsSum[t]);
    atomicAdd(&stats[(b * NGRP + t) * 2 + 1], ldsSq[t]);
  }
}

// ---------------------------------------------------------------------------
// 5) Finalize GroupNorm stats -> (mean, rstd)
// ---------------------------------------------------------------------------
__global__ void gn_finalize_kernel(const float* __restrict__ stats,
                                   float* __restrict__ mr) {
  const int tid = threadIdx.x;               // B*NGRP = 16
  if (tid >= Bsz * NGRP) return;
  const float cnt  = 16.f * (float)NKc;      // 16 channels * N*K
  const float sum  = stats[tid * 2 + 0];
  const float sumq = stats[tid * 2 + 1];
  const float mean = sum / cnt;
  const float var  = sumq / cnt - mean * mean;
  mr[tid * 2 + 0] = mean;
  mr[tid * 2 + 1] = rsqrtf(var + EPSV);
}

// ---------------------------------------------------------------------------
// 6) Apply GN * gamma + beta, LeakyReLU, emit bf16 next-layer input.
//    4-wide vectorized: float4 in, ushort4 (bf16) out.
// ---------------------------------------------------------------------------
__global__ void apply_kernel(const float* __restrict__ Y,
                             const float* __restrict__ mr,
                             const float* __restrict__ gamma,
                             const float* __restrict__ beta,
                             unsigned short* __restrict__ Xout) {
  const unsigned int tid = blockIdx.x * 256u + threadIdx.x;
  if (tid >= (unsigned)Bsz * 128 * NKc / 4) return;
  const unsigned int e0 = tid * 4u;          // layout (b*128+c)*NK + col
  const int c = (e0 >> 17) & 127;
  const int b = e0 >> 24;
  const int g = b * NGRP + (c >> 4);
  const float mean = mr[g * 2 + 0], rstd = mr[g * 2 + 1];
  const float sc = rstd * gamma[c];
  const float sh = beta[c] - mean * sc;
  const float4 y = *(const float4*)&Y[e0];
  float v[4] = {y.x, y.y, y.z, y.w};
  ushort4 o;
  unsigned short* op = (unsigned short*)&o;
#pragma unroll
  for (int i = 0; i < 4; ++i) {
    float x = v[i] * sc + sh;
    x = (x >= 0.f) ? x : SLOPE * x;
    op[i] = f2bf(x);
  }
  *(ushort4*)&Xout[e0] = o;
}

// ---------------------------------------------------------------------------
// 7) Last layer: GN + LeakyReLU fused with max over K -> fp32 output.
// ---------------------------------------------------------------------------
__global__ void apply_max_kernel(const float* __restrict__ Y,
                                 const float* __restrict__ mr,
                                 const float* __restrict__ gamma,
                                 const float* __restrict__ beta,
                                 float* __restrict__ out) {
  const unsigned int tid = blockIdx.x * 256u + threadIdx.x;  // (b*128+c)*N + n
  if (tid >= (unsigned)Bsz * 128 * Npts) return;
  const int c = (tid >> 13) & 127;
  const int b = tid >> 20;
  const int g = b * NGRP + (c >> 4);
  const float mean = mr[g * 2 + 0], rstd = mr[g * 2 + 1];
  const float sc = rstd * gamma[c];
  const float sh = beta[c] - mean * sc;
  const size_t base = (size_t)tid * 16;      // == (b*128+c)*NK + n*16
  float mx = -3.4e38f;
#pragma unroll
  for (int k = 0; k < KNN; ++k) {
    float x = Y[base + k] * sc + sh;
    x = (x >= 0.f) ? x : SLOPE * x;
    mx = fmaxf(mx, x);
  }
  out[tid] = mx;
}

// ---------------------------------------------------------------------------
extern "C" void kernel_launch(void* const* d_in, const int* in_sizes, int n_in,
                              void* d_out, int out_size, void* d_ws, size_t ws_size,
                              hipStream_t stream) {
  (void)in_sizes; (void)n_in; (void)out_size; (void)ws_size;
  const float* xyz1  = (const float*)d_in[0];
  const float* xyz2  = (const float*)d_in[1];
  const float* feat1 = (const float*)d_in[2];
  const float* feat2 = (const float*)d_in[3];
  const float* W[3]  = {(const float*)d_in[4],  (const float*)d_in[8],  (const float*)d_in[12]};
  const float* bia[3]= {(const float*)d_in[5],  (const float*)d_in[9],  (const float*)d_in[13]};
  const float* gam[3]= {(const float*)d_in[6],  (const float*)d_in[10], (const float*)d_in[14]};
  const float* bet[3]= {(const float*)d_in[7],  (const float*)d_in[11], (const float*)d_in[15]};
  float* out = (float*)d_out;
  char* ws = (char*)d_ws;

  // workspace layout
  const size_t off_idx = 0;                                    // B*N*K ints
  const size_t off_w1  = 1u << 20;
  const size_t off_w2  = off_w1 + (size_t)128 * KP1 * 2;
  const size_t off_w3  = off_w2 + (size_t)128 * 128 * 2;
  const size_t off_st  = off_w3 + (size_t)128 * 128 * 2;       // 3 * 32 floats
  const size_t off_mr  = off_st + 3 * 32 * sizeof(float);
  const size_t off_X   = 2u << 20;                             // bf16 activations
  const size_t off_Y   = off_X + (size_t)Bsz * KP1 * NKc * 2;  // fp32 pre-GN

  int*            idx    = (int*)(ws + off_idx);
  unsigned short* Wbf[3] = {(unsigned short*)(ws + off_w1),
                            (unsigned short*)(ws + off_w2),
                            (unsigned short*)(ws + off_w3)};
  float*          st[3]  = {(float*)(ws + off_st),
                            (float*)(ws + off_st) + 32,
                            (float*)(ws + off_st) + 64};
  float*          mr[3]  = {(float*)(ws + off_mr),
                            (float*)(ws + off_mr) + 32,
                            (float*)(ws + off_mr) + 64};
  unsigned short* X = (unsigned short*)(ws + off_X);
  float*          Y = (float*)(ws + off_Y);

  hipMemsetAsync(ws + off_st, 0, 6 * 32 * sizeof(float), stream);

  // KNN
  knn_kernel<<<dim3(Npts / 128, Bsz), 128, 0, stream>>>(xyz1, xyz2, idx);

  // Build layer-1 input (bf16, padded 259 -> 288)
  {
    const unsigned int tot = (unsigned)Bsz * KP1 * NKc;
    build_input_kernel<<<(tot + 255) / 256, 256, 0, stream>>>(
        feat1, feat2, xyz1, xyz2, idx, X);
  }

  // Weights -> bf16
  wconv_kernel<<<(128 * KP1 + 255) / 256, 256, 0, stream>>>(W[0], Wbf[0], CIN1, KP1);
  wconv_kernel<<<(128 * 128 + 255) / 256, 256, 0, stream>>>(W[1], Wbf[1], 128, 128);
  wconv_kernel<<<(128 * 128 + 255) / 256, 256, 0, stream>>>(W[2], Wbf[2], 128, 128);

  const dim3 gGemm(NKc / 64, Bsz);
  const unsigned int totApply4 = (unsigned)Bsz * 128 * NKc / 4;

  // Layer 1
  gemm_gn_kernel<<<gGemm, 256, 0, stream>>>(Wbf[0], X, bia[0], Y, st[0], KP1);
  gn_finalize_kernel<<<1, 32, 0, stream>>>(st[0], mr[0]);
  apply_kernel<<<(totApply4 + 255) / 256, 256, 0, stream>>>(Y, mr[0], gam[0], bet[0], X);

  // Layer 2
  gemm_gn_kernel<<<gGemm, 256, 0, stream>>>(Wbf[1], X, bia[1], Y, st[1], 128);
  gn_finalize_kernel<<<1, 32, 0, stream>>>(st[1], mr[1]);
  apply_kernel<<<(totApply4 + 255) / 256, 256, 0, stream>>>(Y, mr[1], gam[1], bet[1], X);

  // Layer 3 + fused max over K
  gemm_gn_kernel<<<gGemm, 256, 0, stream>>>(Wbf[2], X, bia[2], Y, st[2], 128);
  gn_finalize_kernel<<<1, 32, 0, stream>>>(st[2], mr[2]);
  {
    const unsigned int totOut = (unsigned)Bsz * 128 * Npts;
    apply_max_kernel<<<(totOut + 255) / 256, 256, 0, stream>>>(
        Y, mr[2], gam[2], bet[2], out);
  }
}
